// MultiHeadLSRAttentionFused_63831803953615
// MI455X (gfx1250) — matrix operand order repaired and checked by reading, not compiled, not run on hardware
//
#include <hip/hip_runtime.h>
#include <hip/hip_bf16.h>

// ---------------------------------------------------------------------------
// MultiHeadLSRAttentionFused for MI455X (gfx1250, wave32, WMMA + TDM)
// B=2, T=2048, D=1024, H=16, Dh=64, R=16, causal.
// Pipeline: cvt/transpose(fp32->bf16) -> QKV GEMM (WMMA bf16, double-buffered
//           TDM staging) -> LSR proj -> flash attention (WMMA bf16, online
//           softmax, ds_swizzle reductions) -> out GEMM (WMMA, TDM-staged).
// Workspace: 56 MB -> resident in the 192 MB L2; HBM sees ~50 MB total.
// ---------------------------------------------------------------------------

typedef __bf16 bf16_t;
typedef __bf16 v16bf __attribute__((ext_vector_type(16)));
typedef __bf16 v8bf  __attribute__((ext_vector_type(8)));
typedef __bf16 v4bf  __attribute__((ext_vector_type(4)));
typedef float  v8f   __attribute__((ext_vector_type(8)));
typedef unsigned int u32x4 __attribute__((ext_vector_type(4)));
typedef int i32x4 __attribute__((ext_vector_type(4)));
typedef int i32x8 __attribute__((ext_vector_type(8)));

union Frag16 { v16bf v; v8bf h[2]; };

static __device__ __forceinline__ v8f wmma_bf16(v16bf a, v16bf b, v8f c) {
  return __builtin_amdgcn_wmma_f32_16x16x32_bf16(false, a, false, b,
                                                 (short)0, c, false, false);
}

// ds_swizzle xor-butterfly within 32 lanes: offset15=0, and=0x1f, or=0, xor=m
#define SWZ_XOR(x, m) \
  __int_as_float(__builtin_amdgcn_ds_swizzle(__float_as_int(x), 0x7C00 | (m)))

#define TT 2048
#define HH 16
#define DD 1024
#define DQKV 3072

// ---------------------------------------------------------------------------
// Tensor Data Mover: 2-D tile load (128 rows x 32 bf16) global -> LDS.
// D# built per CDNA5 ISA:
//   group0: count=1 | lds_addr | global_addr[56:0] | type=2
//   group1: data_size=1(2B) | tensor_dim0/1 | tile_dim0=32, tile_dim1=128 |
//           tensor_dim0_stride
// LDS byte address = low 32 bits of the flat pointer (ISA §10.2 LDS aperture).
// ---------------------------------------------------------------------------
static __device__ __forceinline__ void tdm_load_2d(unsigned lds_off,
                                                   const bf16_t* gptr,
                                                   unsigned dim0,     // row len (elems)
                                                   unsigned dim1,     // #rows
                                                   unsigned stride0)  // row stride (elems)
{
  unsigned long long ga = (unsigned long long)(uintptr_t)gptr;
  u32x4 g0;
  g0[0] = 1u;                                                // count=1
  g0[1] = lds_off;                                           // lds_addr
  g0[2] = (unsigned)ga;                                      // global_addr[31:0]
  g0[3] = (unsigned)((ga >> 32) & 0x1FFFFFFu) | (2u << 30);  // [56:32] | type=2
  i32x8 g1;
  g1[0] = (int)(1u << 16);                                   // data_size = 2 bytes
  g1[1] = (int)((dim0 & 0xFFFFu) << 16);                     // tensor_dim0[15:0]
  g1[2] = (int)((dim0 >> 16) | ((dim1 & 0xFFFFu) << 16));    // dim0 hi | dim1 lo
  g1[3] = (int)((dim1 >> 16) | (32u << 16));                 // dim1 hi | tile_dim0=32
  g1[4] = (int)128u;                                         // tile_dim1=128, dim2=0
  g1[5] = (int)stride0;                                      // tensor_dim0_stride lo
  g1[6] = 0;
  g1[7] = 0;
  i32x4 z = {0, 0, 0, 0};
#if __clang_major__ >= 23
  i32x8 z8 = {0, 0, 0, 0, 0, 0, 0, 0};
  __builtin_amdgcn_tensor_load_to_lds(g0, g1, z, z, z8, 0);
#else
  __builtin_amdgcn_tensor_load_to_lds(g0, g1, z, z, 0);
#endif
}

// ---------------------------------------------------------------- cvt fp32->bf16
__global__ __launch_bounds__(256) void cvt_f32_bf16(const float* __restrict__ s,
                                                    bf16_t* __restrict__ d, int n4) {
  int i = blockIdx.x * 256 + threadIdx.x;
  if (i < n4) {
    float4 f = ((const float4*)s)[i];
    v4bf o;
    o.x = (bf16_t)f.x; o.y = (bf16_t)f.y; o.z = (bf16_t)f.z; o.w = (bf16_t)f.w;
    ((v4bf*)d)[i] = o;
  }
}

// ------------------------------------------------- cvt + transpose fp32->bf16
// out[n*Kd + k] = (bf16) in[k*Nd + n]; 32x32 tiles via LDS (coalesced both ways)
__global__ __launch_bounds__(256) void cvt_transpose_bf16(const float* __restrict__ in,
                                                          bf16_t* __restrict__ out,
                                                          int Kd, int Nd) {
  __shared__ float tile[32][33];
  const int k0 = blockIdx.y * 32, n0 = blockIdx.x * 32;
  const int tx = threadIdx.x & 31, ty = threadIdx.x >> 5;  // 32 x 8
#pragma unroll
  for (int r = ty; r < 32; r += 8)
    tile[r][tx] = in[(size_t)(k0 + r) * Nd + n0 + tx];
  __syncthreads();
#pragma unroll
  for (int r = ty; r < 32; r += 8)
    out[(size_t)(n0 + r) * Kd + k0 + tx] = (bf16_t)tile[tx][r];
}

// ---------------------------------------------------------------- bf16 GEMM
// C[M,N] = A[M,K] * Bt[N,K]^T  (Bt stored transposed: row n = K-contiguous).
// Double-buffered TDM pipeline: wave 0 issues the next tile's two DMAs, then
// s_wait_tensorcnt(2) -- TDM ops of one wave complete in order, so <=2
// outstanding means the *current* tile has landed while the next streams in.
// 256 threads = 8 waves; block tile 128x128; wave tile 16x128.
template <bool OUT_F32>
__global__ __launch_bounds__(256) void gemm_bf16(const bf16_t* __restrict__ A,
                                                 const bf16_t* __restrict__ Bt,
                                                 void* __restrict__ C,
                                                 int M, int N, int K) {
  __shared__ __align__(32) bf16_t sA[2][128 * 32];   // (row, k) row-major
  __shared__ __align__(32) bf16_t sBt[2][128 * 32];  // (n, k)  row-major
  const int tid = threadIdx.x;
  const int wv = tid >> 5, lane = tid & 31;
  const int hf = lane >> 4, lm = lane & 15;
  const int m0 = blockIdx.y * 128, n0 = blockIdx.x * 128;

  v8f acc[8] = {};

  const int nk = K / 32;
  if (wv == 0) {  // prologue: DMA tile 0 into buffer 0
    tdm_load_2d((unsigned)(uintptr_t)&sA[0][0], A + (size_t)m0 * K,
                (unsigned)K, (unsigned)M, (unsigned)K);
    tdm_load_2d((unsigned)(uintptr_t)&sBt[0][0], Bt + (size_t)n0 * K,
                (unsigned)K, (unsigned)N, (unsigned)K);
  }

  for (int ki = 0; ki < nk; ++ki) {
    const int cur = ki & 1;
    const int k0 = ki * 32;
    if (wv == 0) {
      if (ki + 1 < nk) {  // stream tile ki+1 while tile ki is consumed
        tdm_load_2d((unsigned)(uintptr_t)&sA[cur ^ 1][0],
                    A + (size_t)m0 * K + k0 + 32, (unsigned)K, (unsigned)M,
                    (unsigned)K);
        tdm_load_2d((unsigned)(uintptr_t)&sBt[cur ^ 1][0],
                    Bt + (size_t)n0 * K + k0 + 32, (unsigned)K, (unsigned)N,
                    (unsigned)K);
        __builtin_amdgcn_s_wait_tensorcnt(2);  // tile ki complete
      } else {
        __builtin_amdgcn_s_wait_tensorcnt(0);
      }
    }
    if (k0 + 64 < K) {  // global_prefetch_b8: warm GL2 for tile ki+2
      __builtin_prefetch(&A[(size_t)(m0 + (tid >> 1)) * K + k0 + 64], 0, 0);
      __builtin_prefetch(&Bt[(size_t)(n0 + (tid >> 1)) * K + k0 + 64], 0, 0);
    }
    __syncthreads();  // publish tile ki to all waves

    // A fragment (ISA 16-bit A 16x32): lane lm, K runs [8h,8h+8) & [16+8h,..)
    Frag16 a;
    const bf16_t* arow = &sA[cur][(wv * 16 + lm) * 32];
    a.h[0] = *(const v8bf*)(arow + 8 * hf);
    a.h[1] = *(const v8bf*)(arow + 16 + 8 * hf);
#pragma unroll
    for (int ns = 0; ns < 8; ++ns) {
      // B fragment: lane lm = column, K runs [8h,8h+8) & [16+8h,..)
      Frag16 b;
      const bf16_t* brow = &sBt[cur][(ns * 16 + lm) * 32];
      b.h[0] = *(const v8bf*)(brow + 8 * hf);
      b.h[1] = *(const v8bf*)(brow + 16 + 8 * hf);
      acc[ns] = wmma_bf16(a.v, b.v, acc[ns]);
    }
    __syncthreads();  // buffer `cur` may be overwritten next iteration
  }

  // C layout: VGPR j -> row j + 8*hf, col = lm
  const int rowb = m0 + wv * 16 + hf * 8;
#pragma unroll
  for (int ns = 0; ns < 8; ++ns) {
    int col = n0 + ns * 16 + lm;
#pragma unroll
    for (int j = 0; j < 8; ++j) {
      if (OUT_F32)
        ((float*)C)[(size_t)(rowb + j) * N + col] = acc[ns][j];
      else
        ((bf16_t*)C)[(size_t)(rowb + j) * N + col] = (bf16_t)acc[ns][j];
    }
  }
}

// ---------------------------------------------------------------- LSR projection
// q_lr[b,h,t,r] = (q . Wq[h,:,r]) * core[h,r] / sqrt(R);  k_lr = k . Wk[h,:,r]
// Stored as (B*H, T, 32) bf16, r=16..31 zero-padded (matches WMMA K=32 frags).
__global__ __launch_bounds__(256) void lsr_proj(const bf16_t* __restrict__ qkvb,
                                                const float* __restrict__ Wq,
                                                const float* __restrict__ Wk,
                                                const float* __restrict__ core,
                                                bf16_t* __restrict__ qlr,
                                                bf16_t* __restrict__ klr) {
  int tid = blockIdx.x * 256 + threadIdx.x;  // B*H*T*R = 2^20 threads
  int r = tid & 15;
  int t = (tid >> 4) & (TT - 1);
  int bh = tid >> 15;          // T*R = 32768
  int b = bh >> 4, h = bh & 15;
  const bf16_t* qrow = qkvb + ((size_t)(b * TT + t)) * DQKV + h * 64;
  const bf16_t* krow = qrow + DD;
  const float* wq = Wq + (h * 64) * 16 + r;
  const float* wk = Wk + (h * 64) * 16 + r;
  float sq = 0.f, sk = 0.f;
#pragma unroll 8
  for (int d = 0; d < 64; ++d) {
    sq += (float)qrow[d] * wq[d * 16];
    sk += (float)krow[d] * wk[d * 16];
  }
  size_t o = ((size_t)bh * TT + t) * 32;
  qlr[o + r]      = (bf16_t)(sq * core[h * 16 + r] * 0.25f);  // 1/sqrt(16)
  qlr[o + 16 + r] = (bf16_t)0.f;
  klr[o + r]      = (bf16_t)sk;
  klr[o + 16 + r] = (bf16_t)0.f;
}

// ---------------------------------------------------------------- flash attention
// grid = (B*H, T/128); 8 waves, each owns a 16-row query tile.
// Per 32-key block: 2 score WMMAs (K=16 real, zero-padded), causal mask,
// online softmax (ds_swizzle xor butterflies within 16-lane halves), P staged
// to per-wave LDS in A-fragment order, V staged transposed, 4 P*V WMMAs.
__global__ __launch_bounds__(256) void lsr_attn(const bf16_t* __restrict__ qlr,
                                                const bf16_t* __restrict__ klr,
                                                const bf16_t* __restrict__ qkvb,
                                                bf16_t* __restrict__ aob) {
  __shared__ __align__(32) bf16_t sVt[64 * 32];      // (d, key) for 32 keys
  __shared__ __align__(32) bf16_t sP[8][16 * 32];    // per-wave P tile (row, key)
  const float NEG_INF = -__builtin_inff();

  const int bh = blockIdx.x;
  const int b = bh >> 4, h = bh & 15;
  const int qt = blockIdx.y;
  const int tid = threadIdx.x;
  const int wv = tid >> 5, lane = tid & 31, hf = lane >> 4, lm = lane & 15;
  const int qb = qt * 128 + wv * 16;

  // q_lr A-fragment: (T,32) row-major already matches the per-lane layout.
  Frag16 aq;
  {
    const bf16_t* qrow = qlr + ((size_t)bh * TT + qb + lm) * 32;
    aq.h[0] = *(const v8bf*)(qrow + 8 * hf);
    aq.h[1] = *(const v8bf*)(qrow + 16 + 8 * hf);
  }

  v8f acc[4] = {};
  float mrow[8], lrow[8];
#pragma unroll
  for (int j = 0; j < 8; ++j) { mrow[j] = NEG_INF; lrow[j] = 0.f; }

  const int nkb = qt * 4 + 4;  // key blocks covering keys 0 .. qt*128+127
  for (int kb = 0; kb < nkb; ++kb) {
    const int k0 = kb * 32;
    __syncthreads();  // previous iteration's sVt consumers are done
    {   // stage V transposed: thread -> (key = tid&31, 8 d's)
      int key = tid & 31, dseg = tid >> 5;
      const bf16_t* vsrc =
          qkvb + ((size_t)(b * TT + k0 + key)) * DQKV + 2 * DD + h * 64 + dseg * 8;
      v8bf vv = *(const v8bf*)vsrc;
#pragma unroll
      for (int u = 0; u < 8; ++u) sVt[(dseg * 8 + u) * 32 + key] = vv[u];
    }
    __syncthreads();

    if (k0 <= qb + 15) {  // wave-uniform: some query row in this wave sees k0
      float p[2][8], tmax[8];
#pragma unroll
      for (int j = 0; j < 8; ++j) tmax[j] = NEG_INF;
#pragma unroll
      for (int sub = 0; sub < 2; ++sub) {
        const int kbase = k0 + sub * 16;
        Frag16 bk;  // k_lr row-major-32 is exactly the B fragment layout
        const bf16_t* krow = klr + ((size_t)bh * TT + kbase + lm) * 32;
        bk.h[0] = *(const v8bf*)(krow + 8 * hf);
        bk.h[1] = *(const v8bf*)(krow + 16 + 8 * hf);
        v8f zero = {};
        v8f c = wmma_bf16(aq.v, bk.v, zero);
        const int keyc = kbase + lm;
#pragma unroll
        for (int j = 0; j < 8; ++j) {
          int rowc = qb + j + 8 * hf;
          float s = c[j];
          if (keyc > rowc) s = NEG_INF;  // causal
          p[sub][j] = s;
          tmax[j] = fmaxf(tmax[j], s);
        }
      }
      // row-wise max across the 16 lanes of each half (ds_swizzle xor mode)
#pragma unroll
      for (int j = 0; j < 8; ++j) {
        float v = tmax[j];
        v = fmaxf(v, SWZ_XOR(v, 1));
        v = fmaxf(v, SWZ_XOR(v, 2));
        v = fmaxf(v, SWZ_XOR(v, 4));
        v = fmaxf(v, SWZ_XOR(v, 8));
        tmax[j] = v;
      }
      float scale[8], rsum[8];
#pragma unroll
      for (int j = 0; j < 8; ++j) {
        float nm = fmaxf(mrow[j], tmax[j]);
        scale[j] = __expf(mrow[j] - nm);  // exp(-inf)=0 on first block
        mrow[j] = nm;
        rsum[j] = 0.f;
      }
#pragma unroll
      for (int sub = 0; sub < 2; ++sub)
#pragma unroll
        for (int j = 0; j < 8; ++j) {
          float e = __expf(p[sub][j] - mrow[j]);  // masked -> exp(-inf)=0
          rsum[j] += e;
          sP[wv][(j + 8 * hf) * 32 + sub * 16 + lm] = (bf16_t)e;
        }
#pragma unroll
      for (int j = 0; j < 8; ++j) {
        float v = rsum[j];
        v += SWZ_XOR(v, 1);
        v += SWZ_XOR(v, 2);
        v += SWZ_XOR(v, 4);
        v += SWZ_XOR(v, 8);
        lrow[j] = lrow[j] * scale[j] + v;
      }
#pragma unroll
      for (int ns = 0; ns < 4; ++ns)
#pragma unroll
        for (int j = 0; j < 8; ++j) acc[ns][j] *= scale[j];

      __builtin_amdgcn_wave_barrier();  // wave-local LDS is in-order (DScnt)

      Frag16 ap;  // P as A fragment (K = 32 keys, all real)
      const bf16_t* prow = &sP[wv][lm * 32];
      ap.h[0] = *(const v8bf*)(prow + 8 * hf);
      ap.h[1] = *(const v8bf*)(prow + 16 + 8 * hf);
#pragma unroll
      for (int ns = 0; ns < 4; ++ns) {
        Frag16 bv;  // V^T rows are d; per-lane K runs [8h,8h+8) & [16+8h,..)
        const bf16_t* vrow = &sVt[(ns * 16 + lm) * 32];
        bv.h[0] = *(const v8bf*)(vrow + 8 * hf);
        bv.h[1] = *(const v8bf*)(vrow + 16 + 8 * hf);
        acc[ns] = wmma_bf16(ap.v, bv.v, acc[ns]);
      }
    }
  }

  // epilogue: normalize by l and scatter to (B*T, D) head-interleaved layout
#pragma unroll
  for (int j = 0; j < 8; ++j) lrow[j] = 1.f / lrow[j];
#pragma unroll
  for (int ns = 0; ns < 4; ++ns)
#pragma unroll
    for (int j = 0; j < 8; ++j) {
      size_t row = (size_t)b * TT + qb + j + 8 * hf;
      aob[row * DD + h * 64 + ns * 16 + lm] = (bf16_t)(acc[ns][j] * lrow[j]);
    }
}

// ---------------------------------------------------------------- launcher
extern "C" void kernel_launch(void* const* d_in, const int* in_sizes, int n_in,
                              void* d_out, int out_size, void* d_ws, size_t ws_size,
                              hipStream_t stream) {
  (void)in_sizes; (void)n_in; (void)out_size; (void)ws_size;
  const float* x      = (const float*)d_in[0];  // (B,T,D)
  const float* W_qkv  = (const float*)d_in[1];  // (D,3D)
  const float* W_qlsr = (const float*)d_in[2];  // (H,Dh,R)
  const float* W_klsr = (const float*)d_in[3];  // (H,Dh,R)
  const float* core   = (const float*)d_in[4];  // (H,R)
  const float* W_o    = (const float*)d_in[5];  // (D,D)
  float* out = (float*)d_out;

  char* ws = (char*)d_ws;                       // 56 MB total
  bf16_t* xb     = (bf16_t*)(ws);               //  8 MB  x bf16
  bf16_t* wqkvbT = (bf16_t*)(ws + (8u  << 20)); //  6 MB  W_qkv^T bf16 (3072,1024)
  bf16_t* wobT   = (bf16_t*)(ws + (14u << 20)); //  2 MB  W_o^T bf16 (1024,1024)
  bf16_t* qkvb   = (bf16_t*)(ws + (16u << 20)); // 24 MB  qkv bf16
  bf16_t* qlr    = (bf16_t*)(ws + (40u << 20)); //  4 MB  q_lr (K=32 padded)
  bf16_t* klr    = (bf16_t*)(ws + (44u << 20)); //  4 MB  k_lr (K=32 padded)
  bf16_t* aob    = (bf16_t*)(ws + (48u << 20)); //  8 MB  attn out bf16

  const int M = 2 * TT;  // 4096 rows

  cvt_f32_bf16<<<(M * DD / 4 + 255) / 256, 256, 0, stream>>>(x, xb, M * DD / 4);
  // weights converted AND transposed so both GEMM operands are K-contiguous
  // (-> both LDS tiles become plain 2-D TDM tile copies)
  cvt_transpose_bf16<<<dim3(DQKV / 32, DD / 32), 256, 0, stream>>>(
      W_qkv, wqkvbT, DD, DQKV);
  cvt_transpose_bf16<<<dim3(DD / 32, DD / 32), 256, 0, stream>>>(
      W_o, wobT, DD, DD);

  // qkv = x @ W_qkv : 4096 x 3072 x 1024 (double-buffered TDM, WMMA bf16)
  gemm_bf16<false><<<dim3(DQKV / 128, M / 128), 256, 0, stream>>>(
      xb, wqkvbT, qkvb, M, DQKV, DD);

  // low-rank projections (with core + 1/sqrt(R) fused into q side)
  lsr_proj<<<(2 * HH * TT * 16) / 256, 256, 0, stream>>>(
      qkvb, W_qlsr, W_klsr, core, qlr, klr);

  // causal flash attention over rank-16 scores
  lsr_attn<<<dim3(2 * HH, TT / 128), 256, 0, stream>>>(qlr, klr, qkvb, aob);

  // final projection: out = attn_out @ W_o : 4096 x 1024 x 1024, fp32 out
  gemm_bf16<true><<<dim3(DD / 128, M / 128), 256, 0, stream>>>(
      aob, wobT, out, M, DD, DD);
}
